// MultiheadAttention_33320356282666
// MI455X (gfx1250) — matrix operand order repaired and checked by reading, not compiled, not run on hardware
//
#include <hip/hip_runtime.h>
#include <hip/hip_bf16.h>
#include <stdint.h>

// ---------------------------------------------------------------------------
// Fused MHA for MI455X (gfx1250, wave32): bf16 WMMA + async-LDS + TDM.
//   B=4, S=2048, E=1024, H=16, DH=64
// ---------------------------------------------------------------------------

typedef __bf16 bf16_t;
typedef bf16_t v16bf __attribute__((ext_vector_type(16)));
typedef float  v8f   __attribute__((ext_vector_type(8)));
typedef uint32_t su4 __attribute__((ext_vector_type(4)));
typedef uint32_t su8 __attribute__((ext_vector_type(8)));

union Frag {
  v16bf v;
  uint32_t d[8];
  uint4 q[2];
  unsigned short u[16];
};

__device__ __forceinline__ unsigned short f2bf(float f) {
  union { float f; uint32_t u; } c; c.f = f;
  uint32_t u = c.u;
  u += 0x7FFFu + ((u >> 16) & 1u);   // round-to-nearest-even
  return (unsigned short)(u >> 16);
}

// k-offset pattern for 16-bit A fragment (16x32): contiguous k = 16*(lane/16) + s
__device__ __forceinline__ int a_k(int j, int koff) {
  return 2 * j + koff;
}

// low 32 bits of a flat shared-memory address == LDS byte offset
__device__ __forceinline__ uint32_t lds_off(const void* p) {
  return (uint32_t)(size_t)p;
}

// async copy of one 128B row (8 x b128 per lane) global -> LDS
__device__ __forceinline__ void async_row128(uint32_t lds, uint64_t ga) {
  asm volatile(
      "global_load_async_to_lds_b128 %0, %1, off\n\t"
      "global_load_async_to_lds_b128 %0, %1, off offset:16\n\t"
      "global_load_async_to_lds_b128 %0, %1, off offset:32\n\t"
      "global_load_async_to_lds_b128 %0, %1, off offset:48\n\t"
      "global_load_async_to_lds_b128 %0, %1, off offset:64\n\t"
      "global_load_async_to_lds_b128 %0, %1, off offset:80\n\t"
      "global_load_async_to_lds_b128 %0, %1, off offset:96\n\t"
      "global_load_async_to_lds_b128 %0, %1, off offset:112"
      :: "v"(lds), "v"(ga) : "memory");
}

__device__ __forceinline__ void wait_async0() {
  asm volatile("s_wait_asynccnt 0x0" ::: "memory");
}

// ---------------------------------------------------------------------------
// Kernel 1: fp32 -> bf16 conversion (grid-stride)
// ---------------------------------------------------------------------------
__global__ void cvt_f32_bf16(const float* __restrict__ src,
                             unsigned short* __restrict__ dst, int n) {
  int i = blockIdx.x * blockDim.x + threadIdx.x;
  int stride = gridDim.x * blockDim.x;
  for (; i < n; i += stride) dst[i] = f2bf(src[i]);
}

// ---------------------------------------------------------------------------
// Kernel 2: QKV projection GEMM
//   [8192,3072] = Xbf[8192,1024] @ Wqkvbf[1024,3072]
//   epilogue scatters bf16 into Q/K/V laid out [B,H,S,64]
// Block: 128 threads = 4 waves, block tile 128x128, wave tile 64x64.
// A tile staged with global_load_async_to_lds_b128 (ASYNCcnt path).
// ---------------------------------------------------------------------------
__global__ __launch_bounds__(128)
void gemm_qkv(const unsigned short* __restrict__ X,
              const unsigned short* __restrict__ W,
              unsigned short* __restrict__ Qb,
              unsigned short* __restrict__ Kb,
              unsigned short* __restrict__ Vb) {
  __shared__ unsigned short As[128][72];   // [m][k], padded (async dst uses [0..63])
  __shared__ unsigned short Bs[128][72];   // [n][k], transposed + padded

  const int tid   = threadIdx.x;
  const int lane  = tid & 31;
  const int wave  = tid >> 5;
  const int waveM = wave >> 1;
  const int waveN = wave & 1;
  const int m0 = blockIdx.y * 128;
  const int n0 = blockIdx.x * 128;

  const int aM    = lane & 15;
  const int aKoff = (lane >> 4) << 4;
  const int bN    = lane & 15;
  const int bKoff = (lane >> 4) << 4;

  const uint32_t ldsA = lds_off(&As[tid][0]);

  v8f acc[4][4] = {};

  for (int kb = 0; kb < 1024; kb += 64) {
    // stage A: thread tid owns row tid, async direct-to-LDS
    async_row128(ldsA, (uint64_t)(size_t)(X + (size_t)(m0 + tid) * 1024 + kb));

    // stage B transposed: coalesced dword reads, 2x u16 LDS writes
    #pragma unroll
    for (int i = 0; i < 32; ++i) {
      int idx = tid + i * 128;
      int k   = idx >> 6;
      int n2  = (idx & 63) << 1;
      uint32_t v = *(const uint32_t*)(W + (size_t)(kb + k) * 3072 + n0 + n2);
      Bs[n2][k]     = (unsigned short)(v & 0xFFFFu);
      Bs[n2 + 1][k] = (unsigned short)(v >> 16);
    }
    wait_async0();
    __syncthreads();

    #pragma unroll
    for (int ks = 0; ks < 2; ++ks) {
      Frag a[4], b[4];
      #pragma unroll
      for (int mi = 0; mi < 4; ++mi) {
        const unsigned short* row = &As[waveM * 64 + mi * 16 + aM][ks * 32];
        #pragma unroll
        for (int j = 0; j < 8; ++j)
          a[mi].d[j] = *(const uint32_t*)&row[a_k(j, aKoff)];
      }
      #pragma unroll
      for (int ni = 0; ni < 4; ++ni) {
        const unsigned short* row = &Bs[waveN * 64 + ni * 16 + bN][ks * 32 + bKoff];
        b[ni].q[0] = *(const uint4*)&row[0];
        b[ni].q[1] = *(const uint4*)&row[8];
      }
      #pragma unroll
      for (int mi = 0; mi < 4; ++mi)
        #pragma unroll
        for (int ni = 0; ni < 4; ++ni)
          acc[mi][ni] = __builtin_amdgcn_wmma_f32_16x16x32_bf16(
              false, a[mi].v, false, b[ni].v, (short)0, acc[mi][ni], false, false);
    }
    __syncthreads();
  }

  // epilogue: scatter into Q/K/V [B,H,S,64] as bf16
  #pragma unroll
  for (int mi = 0; mi < 4; ++mi) {
    #pragma unroll
    for (int ni = 0; ni < 4; ++ni) {
      #pragma unroll
      for (int r = 0; r < 8; ++r) {
        int m = m0 + waveM * 64 + mi * 16 + ((lane >> 4) << 3) + r;
        int n = n0 + waveN * 64 + ni * 16 + (lane & 15);
        int bb = m >> 11;          // batch
        int ss = m & 2047;         // seq
        int h  = n / 192;          // head
        int rr = n - h * 192;      // 0..191 -> q/k/v split
        size_t base = ((size_t)(bb * 16 + h) * 2048 + ss) * 64;
        unsigned short val = f2bf(acc[mi][ni][r]);
        if (rr < 64)       Qb[base + rr]       = val;
        else if (rr < 128) Kb[base + rr - 64]  = val;
        else               Vb[base + rr - 128] = val;
      }
    }
  }
}

// ---------------------------------------------------------------------------
// Kernel 3: flash attention, one (b,h) slice per blockIdx.y, 128 q-rows/block.
// 8 waves x 16 q-rows; 32-key chunks; online softmax; K tile staged by the
// Tensor Data Mover (tensor_load_to_lds + s_wait_tensorcnt), with the TDM
// LDS-padding feature recreating the 72-ushort bank-conflict-free rows.
// ---------------------------------------------------------------------------
__global__ __launch_bounds__(256)
void flash_attn(const unsigned short* __restrict__ Qb,
                const unsigned short* __restrict__ Kb,
                const unsigned short* __restrict__ Vb,
                unsigned short* __restrict__ Val) {
  __shared__ unsigned short Ks[32][72];       // [key][d], padded (TDM pad)
  __shared__ unsigned short Vt[64][40];       // [d][key], transposed + padded
  __shared__ unsigned short Pb[8][16][36];    // per-wave P scratch [m][key]

  const int tid  = threadIdx.x;
  const int lane = tid & 31;
  const int wave = tid >> 5;
  const int bh   = blockIdx.y;
  const size_t hbase = (size_t)bh * 2048 * 64;
  const unsigned short* Qp = Qb + hbase;
  const unsigned short* Kp = Kb + hbase;
  const unsigned short* Vp = Vb + hbase;
  const int q0 = blockIdx.x * 128 + wave * 16;

  const int aM    = lane & 15;
  const int aKoff = (lane >> 4) << 4;
  const int bN    = lane & 15;
  const int bKoff = (lane >> 4) << 4;

  const uint32_t ldsK = lds_off(&Ks[0][0]);

  // Q fragments (held in registers for the whole pass): two 16x32 A-frags
  Frag qf[2];
  #pragma unroll
  for (int ks = 0; ks < 2; ++ks)
    #pragma unroll
    for (int j = 0; j < 8; ++j)
      qf[ks].d[j] =
          *(const uint32_t*)(Qp + (size_t)(q0 + aM) * 64 + ks * 32 + a_k(j, aKoff));

  v8f acc[4] = {};
  float mrow[8], lrow[8], alphar[8];
  #pragma unroll
  for (int r = 0; r < 8; ++r) { mrow[r] = -1e30f; lrow[r] = 0.0f; }

  const float scale = 0.125f;  // 1/sqrt(64)

  // TDM descriptor group1 (chunk-invariant): data_size=2B, pad 4 DWORDs after
  // every 32 DWORDs (row = 64 bf16 -> LDS row stride 72 ushorts),
  // tensor 64 x 2048, tile 64 x 32, dim0 stride 64.
  su8 g1;
  g1[0] = (1u << 16) | (1u << 20) | (4u << 22) | (3u << 25);
  g1[1] = (64u << 16);        // tensor_dim0 = 64
  g1[2] = (2048u << 16);      // tensor_dim1 = 2048 (low 16 in [31:16])
  g1[3] = (64u << 16);        // tensor_dim1 hi = 0; tile_dim0 = 64
  g1[4] = 32u;                // tile_dim1 = 32; tile_dim2 = 0
  g1[5] = 64u;                // tensor_dim0_stride = 64
  g1[6] = 0u;
  g1[7] = 0u;
  su4 gz = {0u, 0u, 0u, 0u};  // groups 2/3 (2D tensor: dims/tiles unused)

  for (int kc = 0; kc < 2048; kc += 32) {
    // stage K chunk (32x64 -> padded 32x72) via Tensor Data Mover, wave 0 only
    if (wave == 0) {
      uint64_t ga = (uint64_t)(size_t)(Kp + (size_t)kc * 64);
      su4 g0;
      g0[0] = 1u;                                   // count=1, user mode
      g0[1] = ldsK;                                 // lds_addr
      g0[2] = (uint32_t)ga;                         // global_addr[31:0]
      g0[3] = ((uint32_t)(ga >> 32) & 0x01FFFFFFu)  // global_addr[56:32]
              | (2u << 30);                         // type=2 (image)
      asm volatile(
          "tensor_load_to_lds %0, %1, %2, %3\n\t"
          "s_wait_tensorcnt 0x0"
          :: "s"(g0), "s"(g1), "s"(gz), "s"(gz) : "memory");
    }
    // stage V chunk transposed -> Vt[d][key]
    #pragma unroll
    for (int i = 0; i < 4; ++i) {
      int idx = tid + i * 256;
      int key = idx >> 5;
      int d2  = (idx & 31) << 1;
      uint32_t v = *(const uint32_t*)(Vp + (size_t)(kc + key) * 64 + d2);
      Vt[d2][key]     = (unsigned short)(v & 0xFFFFu);
      Vt[d2 + 1][key] = (unsigned short)(v >> 16);
    }
    if (kc + 32 < 2048) {
      __builtin_prefetch(Kp + (size_t)(kc + 32) * 64 + lane * 64, 0, 1);
      __builtin_prefetch(Vp + (size_t)(kc + 32) * 64 + lane * 64, 0, 1);
    }
    __syncthreads();

    // scores S = Q . K^T : 16x32 tile = two 16x16 frags, 2 k-steps over DH
    v8f sc[2] = {};
    #pragma unroll
    for (int ks2 = 0; ks2 < 2; ++ks2) {
      #pragma unroll
      for (int nf = 0; nf < 2; ++nf) {
        Frag bk;
        const unsigned short* row = &Ks[nf * 16 + bN][ks2 * 32 + bKoff];
        bk.q[0] = *(const uint4*)&row[0];
        bk.q[1] = *(const uint4*)&row[8];
        sc[nf] = __builtin_amdgcn_wmma_f32_16x16x32_bf16(
            false, qf[ks2].v, false, bk.v, (short)0, sc[nf], false, false);
      }
    }

    // online softmax; row reductions across the 16 lanes of a C-frag group
    #pragma unroll
    for (int r = 0; r < 8; ++r) {
      float s0 = sc[0][r] * scale;
      float s1 = sc[1][r] * scale;
      float t = fmaxf(s0, s1);
      t = fmaxf(t, __shfl_xor(t, 1, 32));
      t = fmaxf(t, __shfl_xor(t, 2, 32));
      t = fmaxf(t, __shfl_xor(t, 4, 32));
      t = fmaxf(t, __shfl_xor(t, 8, 32));
      float mnew = fmaxf(mrow[r], t);
      float al = __expf(mrow[r] - mnew);
      mrow[r] = mnew;
      float p0 = __expf(s0 - mnew);
      float p1 = __expf(s1 - mnew);
      float rs = p0 + p1;
      rs += __shfl_xor(rs, 1, 32);
      rs += __shfl_xor(rs, 2, 32);
      rs += __shfl_xor(rs, 4, 32);
      rs += __shfl_xor(rs, 8, 32);
      lrow[r] = lrow[r] * al + rs;
      alphar[r] = al;
      int row = ((lane >> 4) << 3) + r;
      int col = lane & 15;
      Pb[wave][row][col]      = f2bf(p0);
      Pb[wave][row][col + 16] = f2bf(p1);
    }

    // rescale O accumulators
    #pragma unroll
    for (int nd = 0; nd < 4; ++nd)
      #pragma unroll
      for (int r = 0; r < 8; ++r)
        acc[nd][r] *= alphar[r];

    // reload P in A-fragment layout (same-wave LDS round trip)
    Frag pa;
    #pragma unroll
    for (int j = 0; j < 8; ++j)
      pa.d[j] = *(const uint32_t*)&Pb[wave][aM][a_k(j, aKoff)];

    // O += P . V  (K dim = 32 keys, N = 64 head dims -> 4 WMMAs)
    #pragma unroll
    for (int nd = 0; nd < 4; ++nd) {
      Frag vf;
      const unsigned short* row = &Vt[nd * 16 + bN][bKoff];
      vf.q[0] = *(const uint4*)&row[0];
      vf.q[1] = *(const uint4*)&row[8];
      acc[nd] = __builtin_amdgcn_wmma_f32_16x16x32_bf16(
          false, pa.v, false, vf.v, (short)0, acc[nd], false, false);
    }
    __syncthreads();
  }

  // epilogue: O / l, write bf16 values in [B,S,E] layout for final GEMM
  const int b = bh >> 4;
  const int h = bh & 15;
  #pragma unroll
  for (int r = 0; r < 8; ++r) {
    float inv = 1.0f / lrow[r];
    int s = q0 + ((lane >> 4) << 3) + r;
    #pragma unroll
    for (int nd = 0; nd < 4; ++nd) {
      int d = nd * 16 + (lane & 15);
      Val[(size_t)(b * 2048 + s) * 1024 + h * 64 + d] = f2bf(acc[nd][r] * inv);
    }
  }
}

// ---------------------------------------------------------------------------
// Kernel 4: output projection GEMM
//   Out[8192,1024] (fp32) = Valbf[8192,1024] @ Woutbf[1024,1024]
// ---------------------------------------------------------------------------
__global__ __launch_bounds__(128)
void gemm_out(const unsigned short* __restrict__ X,
              const unsigned short* __restrict__ W,
              float* __restrict__ Out) {
  __shared__ unsigned short As[128][72];
  __shared__ unsigned short Bs[128][72];

  const int tid   = threadIdx.x;
  const int lane  = tid & 31;
  const int wave  = tid >> 5;
  const int waveM = wave >> 1;
  const int waveN = wave & 1;
  const int m0 = blockIdx.y * 128;
  const int n0 = blockIdx.x * 128;

  const int aM    = lane & 15;
  const int aKoff = (lane >> 4) << 4;
  const int bN    = lane & 15;
  const int bKoff = (lane >> 4) << 4;

  const uint32_t ldsA = lds_off(&As[tid][0]);

  v8f acc[4][4] = {};

  for (int kb = 0; kb < 1024; kb += 64) {
    async_row128(ldsA, (uint64_t)(size_t)(X + (size_t)(m0 + tid) * 1024 + kb));

    #pragma unroll
    for (int i = 0; i < 32; ++i) {
      int idx = tid + i * 128;
      int k   = idx >> 6;
      int n2  = (idx & 63) << 1;
      uint32_t v = *(const uint32_t*)(W + (size_t)(kb + k) * 1024 + n0 + n2);
      Bs[n2][k]     = (unsigned short)(v & 0xFFFFu);
      Bs[n2 + 1][k] = (unsigned short)(v >> 16);
    }
    wait_async0();
    __syncthreads();

    #pragma unroll
    for (int ks = 0; ks < 2; ++ks) {
      Frag a[4], b[4];
      #pragma unroll
      for (int mi = 0; mi < 4; ++mi) {
        const unsigned short* row = &As[waveM * 64 + mi * 16 + aM][ks * 32];
        #pragma unroll
        for (int j = 0; j < 8; ++j)
          a[mi].d[j] = *(const uint32_t*)&row[a_k(j, aKoff)];
      }
      #pragma unroll
      for (int ni = 0; ni < 4; ++ni) {
        const unsigned short* row = &Bs[waveN * 64 + ni * 16 + bN][ks * 32 + bKoff];
        b[ni].q[0] = *(const uint4*)&row[0];
        b[ni].q[1] = *(const uint4*)&row[8];
      }
      #pragma unroll
      for (int mi = 0; mi < 4; ++mi)
        #pragma unroll
        for (int ni = 0; ni < 4; ++ni)
          acc[mi][ni] = __builtin_amdgcn_wmma_f32_16x16x32_bf16(
              false, a[mi].v, false, b[ni].v, (short)0, acc[mi][ni], false, false);
    }
    __syncthreads();
  }

  #pragma unroll
  for (int mi = 0; mi < 4; ++mi)
    #pragma unroll
    for (int ni = 0; ni < 4; ++ni)
      #pragma unroll
      for (int r = 0; r < 8; ++r) {
        int m = m0 + waveM * 64 + mi * 16 + ((lane >> 4) << 3) + r;
        int n = n0 + waveN * 64 + ni * 16 + (lane & 15);
        Out[(size_t)m * 1024 + n] = acc[mi][ni][r];
      }
}

// ---------------------------------------------------------------------------
// Launch
// ---------------------------------------------------------------------------
extern "C" void kernel_launch(void* const* d_in, const int* in_sizes, int n_in,
                              void* d_out, int out_size, void* d_ws, size_t ws_size,
                              hipStream_t stream) {
  (void)in_sizes; (void)n_in; (void)out_size; (void)ws_size;
  const float* x    = (const float*)d_in[0];
  const float* Wqkv = (const float*)d_in[1];
  const float* Wout = (const float*)d_in[2];
  float* out = (float*)d_out;

  const size_t NX    = (size_t)4 * 2048 * 1024;  // 8388608
  const size_t NWQKV = (size_t)1024 * 3072;
  const size_t NWOUT = (size_t)1024 * 1024;

  unsigned short* ws   = (unsigned short*)d_ws;
  unsigned short* xbf  = ws;              ws += NX;
  unsigned short* wqb  = ws;              ws += NWQKV;
  unsigned short* wob  = ws;              ws += NWOUT;
  unsigned short* Qb   = ws;              ws += NX;
  unsigned short* Kb   = ws;              ws += NX;
  unsigned short* Vb   = ws;              ws += NX;
  unsigned short* Valb = ws;              ws += NX;

  cvt_f32_bf16<<<dim3(2048), dim3(256), 0, stream>>>(x, xbf, (int)NX);
  cvt_f32_bf16<<<dim3(1024), dim3(256), 0, stream>>>(Wqkv, wqb, (int)NWQKV);
  cvt_f32_bf16<<<dim3(512),  dim3(256), 0, stream>>>(Wout, wob, (int)NWOUT);

  gemm_qkv<<<dim3(24, 64), dim3(128), 0, stream>>>(xbf, wqb, Qb, Kb, Vb);
  flash_attn<<<dim3(16, 64), dim3(256), 0, stream>>>(Qb, Kb, Vb, Valb);
  gemm_out<<<dim3(8, 64), dim3(128), 0, stream>>>(Valb, wob, out);
}
